// PCT_Encoder_90048284327979
// MI455X (gfx1250) — compile-verified, matrix operand order checked by reading, not run on hardware
//
#include <hip/hip_runtime.h>
#include <math.h>

// ---------------------------------------------------------------------------
// Types / helpers
// ---------------------------------------------------------------------------
typedef __bf16 bf16_t;
typedef __attribute__((ext_vector_type(16))) __bf16 bf16x16;
typedef __attribute__((ext_vector_type(8)))  __bf16 bf16x8;
typedef __attribute__((ext_vector_type(8)))  float  f32x8;

__device__ __forceinline__ bf16_t f2bf(float x) {
  union { float f; unsigned u; } v; v.f = x;
  unsigned r = v.u + 0x7FFFu + ((v.u >> 16) & 1u);          // round-to-nearest-even
  unsigned short h = (unsigned short)(r >> 16);
  return __builtin_bit_cast(bf16_t, h);
}
__device__ __forceinline__ float bf2f(bf16_t x) {
  unsigned short h = __builtin_bit_cast(unsigned short, x);
  union { float f; unsigned u; } v; v.u = ((unsigned)h) << 16;
  return v.f;
}

#define EPSBN 1e-5f

// ---------------------------------------------------------------------------
// Core batched bf16 WMMA GEMM:  out[col][m] = epi( sum_k W[m,k] * X[col,k] )
//   W : [M,K]    row-major bf16   (per-batch stride sW elements)
//   X : [COLS,K] row-major bf16   (per-batch stride sX elements)
//   epilogue: y = act(y*scale[m] + shift[m]);  act 0=none 1=relu 2=leaky(0.2)
//   maxpool!=0: atomic fp32 max into outF[col>>5][m] (values must be >= 0)
//
// Block = 8 waves. Waves covering the same 16-column group share one X tile,
// staged into LDS with gfx1250 async global->LDS copies (ASYNCcnt) and
// double-buffered; each wave then runs v_wmma_f32_16x16x32_bf16 per K step.
//   wavesPerCol = min(8, M/16)   (4 when M==64, else 8)
//   colsPerBlock = 8 / wavesPerCol
// Grid: x = colTiles/colsPerBlock, y = mTiles/wavesPerCol, z = batch.
// LDS tile pitch is 80 B/row (16B-aligned, bank-spreading).
// ---------------------------------------------------------------------------
#define XT_PITCH 80
#define XT_TILE  (16 * XT_PITCH)          // 1280 B per 16x32 bf16 tile

__global__ __launch_bounds__(256)
void gemm_bf16_wmma(const bf16_t* __restrict__ W, const bf16_t* __restrict__ X,
                    float* __restrict__ outF, bf16_t* __restrict__ outB,
                    const float* __restrict__ scale, const float* __restrict__ shift,
                    int M, int K, int wavesPerCol, int act, int maxpool,
                    long long sW, long long sX, long long sO)
{
  __shared__ __align__(16) unsigned char xtile[2][2][XT_TILE];  // [buf][cslot][..]

  const int bz = blockIdx.z;
  const bf16_t* Wb = W + (size_t)bz * sW;
  const bf16_t* Xb = X + (size_t)bz * sX;
  const int lane = threadIdx.x & 31;
  const int wave = threadIdx.x >> 5;
  const int colsPerBlock = 8 / wavesPerCol;     // 1 or 2
  const int cslot = wave / wavesPerCol;         // 0 .. colsPerBlock-1
  const int mSub  = wave % wavesPerCol;
  const long long col0 = ((long long)blockIdx.x * colsPerBlock + cslot) << 4;
  const int  m0 = (blockIdx.y * wavesPerCol + mSub) << 4;
  const int  h = lane >> 4;                      // lane half selects K slice
  const int  r = lane & 15;                      // row (A) / column (B)
  const bool issuer = (mSub == 0);

  const bf16_t* aRow = Wb + (size_t)(m0 + r) * K + 8 * h;
  const unsigned ldsBase = (unsigned)(size_t)&xtile[0][0][0];
  // async-stage chunk geometry: 64 chunks of 16B; this lane copies chunk
  // `lane` and `lane+32`; chunk c -> row c>>2, 16B-piece c&3.
  const int c0row = lane >> 2, c0j = lane & 3;
  const int c1row = (lane + 32) >> 2, c1j = lane & 3;   // (lane+32)&3 == lane&3

  f32x8 acc = {0.f, 0.f, 0.f, 0.f, 0.f, 0.f, 0.f, 0.f};

  // ---- prologue: stage k0 = 0 into buffer 0 ----
  if (issuer) {
    unsigned l0 = ldsBase + (unsigned)(cslot * XT_TILE + c0row * XT_PITCH + c0j * 16);
    unsigned l1 = ldsBase + (unsigned)(cslot * XT_TILE + c1row * XT_PITCH + c1j * 16);
    unsigned long long g0 = (unsigned long long)(size_t)(Xb + (size_t)(col0 + c0row) * K + c0j * 8);
    unsigned long long g1 = (unsigned long long)(size_t)(Xb + (size_t)(col0 + c1row) * K + c1j * 8);
    asm volatile("global_load_async_to_lds_b128 %0, %1, off" :: "v"(l0), "v"(g0) : "memory");
    asm volatile("global_load_async_to_lds_b128 %0, %1, off" :: "v"(l1), "v"(g1) : "memory");
  }

  int buf = 0;
  for (int k0 = 0; k0 < K; k0 += 32, buf ^= 1) {
    const bool hasNext = (k0 + 32 < K);
    if (issuer) {
      if (hasNext) {                              // stage next K tile into other buffer
        int nb = buf ^ 1;
        unsigned l0 = ldsBase + (unsigned)(nb * 2 * XT_TILE + cslot * XT_TILE + c0row * XT_PITCH + c0j * 16);
        unsigned l1 = ldsBase + (unsigned)(nb * 2 * XT_TILE + cslot * XT_TILE + c1row * XT_PITCH + c1j * 16);
        unsigned long long g0 = (unsigned long long)(size_t)(Xb + (size_t)(col0 + c0row) * K + k0 + 32 + c0j * 8);
        unsigned long long g1 = (unsigned long long)(size_t)(Xb + (size_t)(col0 + c1row) * K + k0 + 32 + c1j * 8);
        asm volatile("global_load_async_to_lds_b128 %0, %1, off" :: "v"(l0), "v"(g0) : "memory");
        asm volatile("global_load_async_to_lds_b128 %0, %1, off" :: "v"(l1), "v"(g1) : "memory");
        asm volatile("s_wait_asynccnt 2" ::: "memory");   // current buffer complete
      } else {
        asm volatile("s_wait_asynccnt 0" ::: "memory");
      }
    }
    __syncthreads();                              // publish LDS tile to all waves

    // A fragment: two 16B chunks straight from global (small, cache-resident)
    bf16x8 alo = *(const bf16x8*)(aRow + k0);
    bf16x8 ahi = *(const bf16x8*)(aRow + k0 + 16);
    if (hasNext) __builtin_prefetch(aRow + k0 + 32, 0, 1);   // global_prefetch_b8
    // B fragment from LDS (ds_load_b128 x2)
    const unsigned char* tb = &xtile[buf][cslot][0] + r * XT_PITCH + h * 16;
    bf16x8 blo = *(const bf16x8*)(tb);
    bf16x8 bhi = *(const bf16x8*)(tb + 32);
    bf16x16 afrag = __builtin_shufflevector(alo, ahi, 0,1,2,3,4,5,6,7,8,9,10,11,12,13,14,15);
    bf16x16 bfrag = __builtin_shufflevector(blo, bhi, 0,1,2,3,4,5,6,7,8,9,10,11,12,13,14,15);
    acc = __builtin_amdgcn_wmma_f32_16x16x32_bf16(false, afrag, false, bfrag,
                                                  (short)0, acc, false, false);
    __syncthreads();                              // consumers done before overwrite
  }

  const long long col = col0 + r;
  const int mbase = m0 + 8 * h;
  float res[8];
#pragma unroll
  for (int v = 0; v < 8; ++v) {
    int m = mbase + v;
    float y = acc[v];
    float s = scale ? scale[m] : 1.f;
    float t = shift ? shift[m] : 0.f;
    y = y * s + t;
    if (act == 1)      y = fmaxf(y, 0.f);
    else if (act == 2) y = (y > 0.f) ? y : 0.2f * y;
    res[v] = y;
  }
  if (maxpool) {
    unsigned* dst = (unsigned*)(outF + (size_t)bz * sO + (size_t)(col >> 5) * M + mbase);
#pragma unroll
    for (int v = 0; v < 8; ++v) atomicMax(&dst[v], __float_as_uint(res[v]));
  } else {
    if (outF) {
      float* dst = outF + (size_t)bz * sO + (size_t)col * M + mbase;
#pragma unroll
      for (int v = 0; v < 8; ++v) dst[v] = res[v];
    }
    if (outB) {
      bf16_t* dst = outB + (size_t)bz * sO + (size_t)col * M + mbase;
#pragma unroll
      for (int v = 0; v < 8; ++v) dst[v] = f2bf(res[v]);
    }
  }
}

// ---------------------------------------------------------------------------
// Small prep / elementwise kernels
// ---------------------------------------------------------------------------
__global__ void prep_ss(const float* __restrict__ g, const float* __restrict__ bnb,
                        const float* __restrict__ bias,
                        float* __restrict__ scale, float* __restrict__ shift, int M)
{
  int i = blockIdx.x * blockDim.x + threadIdx.x;
  if (i >= M) return;
  float s = g ? g[i] * rsqrtf(1.f + EPSBN) : 1.f;
  float t = (bias ? bias[i] * s : 0.f) + (bnb ? bnb[i] : 0.f);
  scale[i] = s;  shift[i] = t;
}

__global__ void cvt_f32_bf16(const float* __restrict__ in, bf16_t* __restrict__ out, long long n)
{
  long long t = (long long)blockIdx.x * blockDim.x + threadIdx.x;
  if (t < n) out[t] = f2bf(in[t]);
}

__global__ void fill_zero_f32(float* __restrict__ p, long long n)
{
  long long t = (long long)blockIdx.x * blockDim.x + threadIdx.x;
  if (t < n) p[t] = 0.f;
}

// in[cols][3] * W[M,3]  (+ bn-style epilogue).  Used for the 3->64 mlp and 3->256 pos.
__global__ void conv3_kernel(const float* __restrict__ in, const float* __restrict__ W,
                             const float* __restrict__ scale, const float* __restrict__ shift,
                             float* __restrict__ outF, bf16_t* __restrict__ outB,
                             long long cols, int M, int act)
{
  long long t = (long long)blockIdx.x * blockDim.x + threadIdx.x;
  if (t >= cols * M) return;
  int m = (int)(t % M);
  long long col = t / M;
  const float* xi = in + col * 3;
  float y = W[m * 3] * xi[0] + W[m * 3 + 1] * xi[1] + W[m * 3 + 2] * xi[2];
  float s = scale ? scale[m] : 1.f;
  float sh = shift ? shift[m] : 0.f;
  y = y * s + sh;
  if (act == 1)      y = fmaxf(y, 0.f);
  else if (act == 2) y = (y > 0.f) ? y : 0.2f * y;
  if (outF) outF[col * M + m] = y;
  if (outB) outB[col * M + m] = f2bf(y);
}

__global__ void ew_add2(const float* __restrict__ a, const float* __restrict__ b,
                        float* __restrict__ outF, bf16_t* __restrict__ outB, long long n)
{
  long long t = (long long)blockIdx.x * blockDim.x + threadIdx.x;
  if (t >= n) return;
  float v = a[t] + b[t];
  if (outF) outF[t] = v;
  if (outB) outB[t] = f2bf(v);
}

__global__ void ew_sub_bf(const float* __restrict__ a, const float* __restrict__ b,
                          bf16_t* __restrict__ out, long long n)
{
  long long t = (long long)blockIdx.x * blockDim.x + threadIdx.x;
  if (t < n) out[t] = f2bf(a[t] - b[t]);
}

// ---------------------------------------------------------------------------
// Farthest point sampling: one workgroup per batch, sequential S steps.
// ---------------------------------------------------------------------------
__global__ __launch_bounds__(256)
void fps_kernel(const float* __restrict__ xyz, int Npts, int S, int* __restrict__ outIdx)
{
  __shared__ float dist[2048];
  __shared__ float sval[256];
  __shared__ int   sidx[256];
  __shared__ int   sFar;
  const int b = blockIdx.x;
  const float* px = xyz + (size_t)b * Npts * 3;
  const int tid = threadIdx.x;
  for (int n = tid; n < Npts; n += 256) dist[n] = 1e10f;
  if (tid == 0) sFar = 0;
  __syncthreads();
  for (int s = 0; s < S; ++s) {
    int far = sFar;
    if (tid == 0) outIdx[b * S + s] = far;
    float cx = px[far * 3], cy = px[far * 3 + 1], cz = px[far * 3 + 2];
    float best = -1.f; int bestIdx = 0x7FFFFFFF;
    for (int n = tid; n < Npts; n += 256) {
      float dx = px[n * 3] - cx, dy = px[n * 3 + 1] - cy, dz = px[n * 3 + 2] - cz;
      float d = fminf(dist[n], dx * dx + dy * dy + dz * dz);
      dist[n] = d;
      if (d > best) { best = d; bestIdx = n; }
    }
    sval[tid] = best; sidx[tid] = bestIdx;
    __syncthreads();
    for (int off = 128; off > 0; off >>= 1) {
      if (tid < off) {
        float v2 = sval[tid + off]; int i2 = sidx[tid + off];
        if (v2 > sval[tid] || (v2 == sval[tid] && i2 < sidx[tid])) { sval[tid] = v2; sidx[tid] = i2; }
      }
      __syncthreads();
    }
    if (tid == 0) sFar = sidx[0];
    __syncthreads();
  }
}

// ---------------------------------------------------------------------------
// KNN (k smallest squared distances, ties -> lowest index, ascending order).
// One workgroup per (b, s) query.
// ---------------------------------------------------------------------------
__global__ __launch_bounds__(256)
void knn_kernel(const float* __restrict__ xyz, const float* __restrict__ qxyz,
                int Npts, int S, int Kn, int* __restrict__ outIdx)
{
  __shared__ float d[2048];
  __shared__ float sval[256];
  __shared__ int   sidx[256];
  const int b = blockIdx.y, s = blockIdx.x;
  const float* px = xyz + (size_t)b * Npts * 3;
  const float* q  = qxyz + ((size_t)b * S + s) * 3;
  const float cx = q[0], cy = q[1], cz = q[2];
  const int tid = threadIdx.x;
  for (int n = tid; n < Npts; n += 256) {
    float dx = px[n * 3] - cx, dy = px[n * 3 + 1] - cy, dz = px[n * 3 + 2] - cz;
    d[n] = dx * dx + dy * dy + dz * dz;
  }
  __syncthreads();
  for (int j = 0; j < Kn; ++j) {
    float best = 3.4e38f; int bestIdx = 0x7FFFFFFF;
    for (int n = tid; n < Npts; n += 256)
      if (d[n] < best) { best = d[n]; bestIdx = n; }
    sval[tid] = best; sidx[tid] = bestIdx;
    __syncthreads();
    for (int off = 128; off > 0; off >>= 1) {
      if (tid < off) {
        float v2 = sval[tid + off]; int i2 = sidx[tid + off];
        if (v2 < sval[tid] || (v2 == sval[tid] && i2 < sidx[tid])) { sval[tid] = v2; sidx[tid] = i2; }
      }
      __syncthreads();
    }
    if (tid == 0) { outIdx[((size_t)(b * S + s)) * Kn + j] = sidx[0]; d[sidx[0]] = 3.4e38f; }
    __syncthreads();
  }
}

// ---------------------------------------------------------------------------
// Gathers + group/concat
// ---------------------------------------------------------------------------
__global__ void gather_xyz(const float* __restrict__ xyz, const int* __restrict__ idx,
                           float* __restrict__ out, int Npts, int S)
{
  long long t = (long long)blockIdx.x * blockDim.x + threadIdx.x;
  int b = blockIdx.y;
  if (t >= (long long)S * 3) return;
  int c = (int)(t % 3);
  int s = (int)(t / 3);
  int id = idx[b * S + s];
  out[((size_t)b * S + s) * 3 + c] = xyz[((size_t)b * Npts + id) * 3 + c];
}

__global__ void gather_feat_bf(const bf16_t* __restrict__ pts, const int* __restrict__ idx,
                               bf16_t* __restrict__ out, int Npts, int S, int D)
{
  long long t = (long long)blockIdx.x * blockDim.x + threadIdx.x;
  if (t >= (long long)S * D) return;
  int b = blockIdx.y;
  int c = (int)(t % D);
  int s = (int)(t / D);
  int id = idx[b * S + s];
  out[((size_t)b * S + s) * D + c] = pts[((size_t)b * Npts + id) * D + c];
}

// feat[col = (b*S+s)*32 + k][0:D]   = pts[b][knn][c] - newpts[b][s][c]
// feat[col][D:2D]                   = newpts[b][s][c]
__global__ void group_concat(const bf16_t* __restrict__ pts, const bf16_t* __restrict__ newpts,
                             const int* __restrict__ knnIdx, bf16_t* __restrict__ feat,
                             int Npts, int S, int D, long long total)
{
  long long t = (long long)blockIdx.x * blockDim.x + threadIdx.x;
  if (t >= total) return;
  int c = (int)(t % D);
  long long colg = t / D;
  int k = (int)(colg & 31);
  long long bs = colg >> 5;
  int b = (int)(bs / S);
  int nb = knnIdx[bs * 32 + k];
  float gp = bf2f(pts[((size_t)b * Npts + nb) * D + c]);
  float np = bf2f(newpts[(size_t)bs * D + c]);
  feat[colg * (2 * D) + c]     = f2bf(gp - np);
  feat[colg * (2 * D) + D + c] = f2bf(np);
}

// ---------------------------------------------------------------------------
// Attention helpers
// ---------------------------------------------------------------------------
__global__ __launch_bounds__(256)
void softmax_rows256(float* __restrict__ e)   // rows of length 256, one block per row
{
  __shared__ float red[256];
  float* p = e + (size_t)blockIdx.x * 256;
  int tid = threadIdx.x;
  float v = p[tid];
  red[tid] = v; __syncthreads();
  for (int off = 128; off > 0; off >>= 1) {
    if (tid < off) red[tid] = fmaxf(red[tid], red[tid + off]);
    __syncthreads();
  }
  float mx = red[0]; __syncthreads();
  float ex = __expf(v - mx);
  red[tid] = ex; __syncthreads();
  for (int off = 128; off > 0; off >>= 1) {
    if (tid < off) red[tid] += red[tid + off];
    __syncthreads();
  }
  p[tid] = ex / red[0];
}

// colsum over n, renormalize, write transposed bf16:  attnT[b][m][n]
__global__ __launch_bounds__(256)
void attn_finalize(const float* __restrict__ attn, bf16_t* __restrict__ attnT)
{
  __shared__ float red[256];
  int b = blockIdx.y, m = blockIdx.x, n = threadIdx.x;
  const float* base = attn + (size_t)b * 256 * 256;
  float v = base[(size_t)n * 256 + m];
  red[n] = v; __syncthreads();
  for (int off = 128; off > 0; off >>= 1) {
    if (n < off) red[n] += red[n + off];
    __syncthreads();
  }
  attnT[((size_t)b * 256 + m) * 256 + n] = f2bf(v / (1e-9f + red[0]));
}

__global__ void transpose_bf(const bf16_t* __restrict__ in, bf16_t* __restrict__ out, int R, int C)
{
  long long t = (long long)blockIdx.x * blockDim.x + threadIdx.x;
  if (t >= (long long)R * C) return;
  int b = blockIdx.y;
  int c = (int)(t % C);
  int r = (int)(t / C);
  out[((size_t)b * C + c) * R + r] = in[((size_t)b * R + r) * C + c];
}

// concat [x1,x2,x3,x4,f1] along channel -> bf16 [col][1280]
__global__ void concat_cat(const float* __restrict__ x1, const float* __restrict__ x2,
                           const float* __restrict__ x3, const float* __restrict__ x4,
                           const float* __restrict__ f1, bf16_t* __restrict__ cat, long long cols)
{
  long long t = (long long)blockIdx.x * blockDim.x + threadIdx.x;
  if (t >= cols * 1280) return;
  int ch = (int)(t % 1280);
  long long col = t / 1280;
  float v;
  if      (ch < 256)  v = x1[col * 256 + ch];
  else if (ch < 512)  v = x2[col * 256 + ch - 256];
  else if (ch < 768)  v = x3[col * 256 + ch - 512];
  else if (ch < 1024) v = x4[col * 256 + ch - 768];
  else                v = f1[col * 256 + ch - 1024];
  cat[t] = f2bf(v);
}

// g[b][m] = max over pts of fused[(b*pts+p)*M + m], written bf16
__global__ void rowmax_pts(const float* __restrict__ fused, bf16_t* __restrict__ g,
                           int pts, int M, long long total)
{
  long long t = (long long)blockIdx.x * blockDim.x + threadIdx.x;
  if (t >= total) return;
  int m = (int)(t % M);
  int b = (int)(t / M);
  float mx = -3.4e38f;
  for (int p = 0; p < pts; ++p)
    mx = fmaxf(mx, fused[((size_t)b * pts + p) * M + m]);
  g[t] = f2bf(mx);
}

// ---------------------------------------------------------------------------
// Host-side orchestration
// ---------------------------------------------------------------------------
extern "C" void kernel_launch(void* const* d_in, const int* in_sizes, int n_in,
                              void* d_out, int out_size, void* d_ws, size_t ws_size,
                              hipStream_t stream)
{
  (void)in_sizes; (void)n_in; (void)out_size; (void)ws_size;
  enum { Bn = 32, Np = 2048, S1 = 512, S2 = 256 };
  const long long COLS0 = (long long)Bn * Np;        // 65536
  const long long COLS1 = (long long)Bn * S1 * 32;   // 524288
  const long long COLS2 = (long long)Bn * S2 * 32;   // 262144
  const long long PTS2  = (long long)Bn * S2;        // 8192

  // ---- input pointers (setup_inputs dict order) ----
  const float* x       = (const float*)d_in[0];
  const float* mlp_W1  = (const float*)d_in[1];
  const float* mlp_g1  = (const float*)d_in[2];
  const float* mlp_b1  = (const float*)d_in[3];
  const float* mlp_W2  = (const float*)d_in[4];
  const float* mlp_g2  = (const float*)d_in[5];
  const float* mlp_b2  = (const float*)d_in[6];
  const float* lo0_W1  = (const float*)d_in[7];
  const float* lo0_g1  = (const float*)d_in[8];
  const float* lo0_b1  = (const float*)d_in[9];
  const float* lo0_W2  = (const float*)d_in[10];
  const float* lo0_g2  = (const float*)d_in[11];
  const float* lo0_b2  = (const float*)d_in[12];
  const float* lo1_W1  = (const float*)d_in[13];
  const float* lo1_g1  = (const float*)d_in[14];
  const float* lo1_b1  = (const float*)d_in[15];
  const float* lo1_W2  = (const float*)d_in[16];
  const float* lo1_g2  = (const float*)d_in[17];
  const float* lo1_b2  = (const float*)d_in[18];
  const float* pt_W1   = (const float*)d_in[19];
  const float* pt_g1   = (const float*)d_in[20];
  const float* pt_b1   = (const float*)d_in[21];
  const float* pos_W   = (const float*)d_in[22];
  const float* pos_b   = (const float*)d_in[23];
  const float* oa_qkW[4], *oa_qkb[4], *oa_vW[4], *oa_vb[4],
             * oa_trW[4], *oa_trb[4], *oa_bng[4], *oa_bnb[4];
  for (int i = 0; i < 4; ++i) {
    int base = 24 + 8 * i;
    oa_qkW[i] = (const float*)d_in[base + 0];
    oa_qkb[i] = (const float*)d_in[base + 1];
    oa_vW[i]  = (const float*)d_in[base + 2];
    oa_vb[i]  = (const float*)d_in[base + 3];
    oa_trW[i] = (const float*)d_in[base + 4];
    oa_trb[i] = (const float*)d_in[base + 5];
    oa_bng[i] = (const float*)d_in[base + 6];
    oa_bnb[i] = (const float*)d_in[base + 7];
  }
  const float* fuse_W = (const float*)d_in[56];
  const float* fuse_g = (const float*)d_in[57];
  const float* fuse_b = (const float*)d_in[58];
  const float* mlp2_W = (const float*)d_in[59];
  const float* mlp2_g = (const float*)d_in[60];
  const float* mlp2_b = (const float*)d_in[61];
  float* outp = (float*)d_out;

  // ---- workspace bump allocator ----
  char* base_ws = (char*)d_ws;
  size_t off = 0;
  auto alloc = [&](size_t bytes) -> void* {
    void* p = base_ws + off;
    off += (bytes + 255) & ~(size_t)255;
    return p;
  };
  bf16_t* h1_bf    = (bf16_t*)alloc(COLS0 * 64 * 2);
  bf16_t* pts_bf   = (bf16_t*)alloc(COLS0 * 64 * 2);
  int*    fps1     = (int*)   alloc(Bn * S1 * 4);
  float*  nxyz1    = (float*) alloc(Bn * S1 * 3 * 4);
  bf16_t* npts1    = (bf16_t*)alloc(Bn * S1 * 64 * 2);
  int*    knn1     = (int*)   alloc((size_t)Bn * S1 * 32 * 4);
  bf16_t* bigA     = (bf16_t*)alloc((size_t)COLS1 * 128 * 2);   // 134 MB
  bf16_t* bigB     = (bf16_t*)alloc((size_t)COLS1 * 128 * 2);   // 134 MB
  float*  f0f      = (float*) alloc((size_t)Bn * S1 * 128 * 4);
  bf16_t* f0bf     = (bf16_t*)alloc((size_t)Bn * S1 * 128 * 2);
  int*    fps2     = (int*)   alloc(Bn * S2 * 4);
  float*  nxyz2    = (float*) alloc(Bn * S2 * 3 * 4);
  bf16_t* npts2    = (bf16_t*)alloc((size_t)Bn * S2 * 128 * 2);
  int*    knn2     = (int*)   alloc((size_t)Bn * S2 * 32 * 4);
  float*  f1f      = (float*) alloc(PTS2 * 256 * 4);
  bf16_t* f1bf     = (bf16_t*)alloc(PTS2 * 256 * 2);
  float*  posb     = (float*) alloc(PTS2 * 256 * 4);
  float*  cur0     = (float*) alloc(PTS2 * 256 * 4);
  float*  xin      = (float*) alloc(PTS2 * 256 * 4);
  bf16_t* xin_bf   = (bf16_t*)alloc(PTS2 * 256 * 2);
  bf16_t* q_bf     = (bf16_t*)alloc(PTS2 * 64 * 2);
  bf16_t* v_bf     = (bf16_t*)alloc(PTS2 * 256 * 2);
  bf16_t* vT_bf    = (bf16_t*)alloc(PTS2 * 256 * 2);
  float*  energy   = (float*) alloc(PTS2 * 256 * 4);
  bf16_t* attnT    = (bf16_t*)alloc(PTS2 * 256 * 2);
  float*  xr       = (float*) alloc(PTS2 * 256 * 4);
  bf16_t* d_bf     = (bf16_t*)alloc(PTS2 * 256 * 2);
  float*  tbuf     = (float*) alloc(PTS2 * 256 * 4);
  float*  xbuf[4];
  for (int i = 0; i < 4; ++i) xbuf[i] = (float*)alloc(PTS2 * 256 * 4);
  bf16_t* cat_bf   = (bf16_t*)alloc(PTS2 * 1280 * 2);
  float*  fused    = (float*) alloc(PTS2 * 1024 * 4);
  bf16_t* g_bf     = (bf16_t*)alloc(Bn * 1024 * 2);
  // bf16 weights
  bf16_t* w_mlp2c  = (bf16_t*)alloc(64 * 64 * 2);
  bf16_t* w_lo0a   = (bf16_t*)alloc(128 * 128 * 2);
  bf16_t* w_lo0b   = (bf16_t*)alloc(128 * 128 * 2);
  bf16_t* w_lo1a   = (bf16_t*)alloc(256 * 256 * 2);
  bf16_t* w_lo1b   = (bf16_t*)alloc(256 * 256 * 2);
  bf16_t* w_pt     = (bf16_t*)alloc(256 * 256 * 2);
  bf16_t* w_qk[4], *w_v[4], *w_tr[4];
  for (int i = 0; i < 4; ++i) {
    w_qk[i] = (bf16_t*)alloc(64 * 256 * 2);
    w_v[i]  = (bf16_t*)alloc(256 * 256 * 2);
    w_tr[i] = (bf16_t*)alloc(256 * 256 * 2);
  }
  bf16_t* w_fuse   = (bf16_t*)alloc((size_t)1024 * 1280 * 2);
  bf16_t* w_mlp2   = (bf16_t*)alloc((size_t)512 * 1024 * 2);
  // fused scale/shift pairs
  auto allocSS = [&](int M) { return (float*)alloc((size_t)M * 2 * 4); };
  float* ss_mlp1 = allocSS(64),  *ss_mlp2c = allocSS(64);
  float* ss_lo0a = allocSS(128), *ss_lo0b  = allocSS(128);
  float* ss_lo1a = allocSS(256), *ss_lo1b  = allocSS(256);
  float* ss_pt   = allocSS(256);
  float* ss_q[4], *ss_v[4], *ss_tr[4];
  for (int i = 0; i < 4; ++i) { ss_q[i] = allocSS(64); ss_v[i] = allocSS(256); ss_tr[i] = allocSS(256); }
  float* ss_fuse = allocSS(1024), *ss_fin = allocSS(512);

  // ---- launch helpers ----
  auto g1d = [](long long n) { return dim3((unsigned)((n + 255) >> 8)); };
  auto cvtW = [&](const float* src, bf16_t* dst, long long n) {
    cvt_f32_bf16<<<g1d(n), 256, 0, stream>>>(src, dst, n);
  };
  auto prep = [&](const float* g, const float* bnb, const float* bias, float* ss, int M) {
    prep_ss<<<g1d(M), 256, 0, stream>>>(g, bnb, bias, ss, ss + M, M);
  };
  auto gemm = [&](const bf16_t* W, const bf16_t* X, float* oF, bf16_t* oB,
                  const float* ss, int M, int K, long long COLS, int act, int maxpool,
                  long long sW, long long sX, long long sO, int batches) {
    int mTiles = M >> 4;
    int wavesPerCol = (mTiles >= 8) ? 8 : mTiles;     // 4 when M==64, else 8
    int colsPerBlock = 8 / wavesPerCol;
    dim3 grid((unsigned)((COLS >> 4) / colsPerBlock), (unsigned)(mTiles / wavesPerCol), batches);
    gemm_bf16_wmma<<<grid, 256, 0, stream>>>(W, X, oF, oB,
        ss ? ss : nullptr, ss ? ss + M : nullptr, M, K, wavesPerCol, act, maxpool, sW, sX, sO);
  };

  // ---- weight conversion + scale/shift prep ----
  cvtW(mlp_W2, w_mlp2c, 64 * 64);
  cvtW(lo0_W1, w_lo0a, 128 * 128);  cvtW(lo0_W2, w_lo0b, 128 * 128);
  cvtW(lo1_W1, w_lo1a, 256 * 256);  cvtW(lo1_W2, w_lo1b, 256 * 256);
  cvtW(pt_W1,  w_pt,   256 * 256);
  for (int i = 0; i < 4; ++i) {
    cvtW(oa_qkW[i], w_qk[i], 64 * 256);
    cvtW(oa_vW[i],  w_v[i],  256 * 256);
    cvtW(oa_trW[i], w_tr[i], 256 * 256);
  }
  cvtW(fuse_W, w_fuse, (long long)1024 * 1280);
  cvtW(mlp2_W, w_mlp2, (long long)512 * 1024);

  prep(mlp_g1, mlp_b1, nullptr, ss_mlp1, 64);
  prep(mlp_g2, mlp_b2, nullptr, ss_mlp2c, 64);
  prep(lo0_g1, lo0_b1, nullptr, ss_lo0a, 128);
  prep(lo0_g2, lo0_b2, nullptr, ss_lo0b, 128);
  prep(lo1_g1, lo1_b1, nullptr, ss_lo1a, 256);
  prep(lo1_g2, lo1_b2, nullptr, ss_lo1b, 256);
  prep(pt_g1,  pt_b1,  nullptr, ss_pt, 256);
  for (int i = 0; i < 4; ++i) {
    prep(nullptr, nullptr, oa_qkb[i], ss_q[i], 64);
    prep(nullptr, nullptr, oa_vb[i],  ss_v[i], 256);
    prep(oa_bng[i], oa_bnb[i], oa_trb[i], ss_tr[i], 256);
  }
  prep(fuse_g, fuse_b, nullptr, ss_fuse, 1024);
  prep(mlp2_g, mlp2_b, nullptr, ss_fin, 512);

  // ---- input MLP: 3 -> 64 (relu bn), 64 -> 64 (relu bn) ----
  conv3_kernel<<<g1d(COLS0 * 64), 256, 0, stream>>>(x, mlp_W1, ss_mlp1, ss_mlp1 + 64,
                                                    nullptr, h1_bf, COLS0, 64, 1);
  gemm(w_mlp2c, h1_bf, nullptr, pts_bf, ss_mlp2c, 64, 64, COLS0, 1, 0, 0, 0, 0, 1);

  // ---- stage 1: sample_and_group(512, 32) + local_op(128) ----
  fps_kernel<<<Bn, 256, 0, stream>>>(x, Np, S1, fps1);
  gather_xyz<<<dim3((unsigned)((S1 * 3 + 255) / 256), Bn), 256, 0, stream>>>(x, fps1, nxyz1, Np, S1);
  gather_feat_bf<<<dim3((unsigned)((S1 * 64 + 255) / 256), Bn), 256, 0, stream>>>(pts_bf, fps1, npts1, Np, S1, 64);
  knn_kernel<<<dim3(S1, Bn), 256, 0, stream>>>(x, nxyz1, Np, S1, 32, knn1);
  group_concat<<<g1d(COLS1 * 64), 256, 0, stream>>>(pts_bf, npts1, knn1, bigA, Np, S1, 64, COLS1 * 64);
  gemm(w_lo0a, bigA, nullptr, bigB, ss_lo0a, 128, 128, COLS1, 1, 0, 0, 0, 0, 1);
  fill_zero_f32<<<g1d((long long)Bn * S1 * 128), 256, 0, stream>>>(f0f, (long long)Bn * S1 * 128);
  gemm(w_lo0b, bigB, f0f, nullptr, ss_lo0b, 128, 128, COLS1, 1, /*maxpool*/1, 0, 0, 0, 1);
  cvt_f32_bf16<<<g1d((long long)Bn * S1 * 128), 256, 0, stream>>>(f0f, f0bf, (long long)Bn * S1 * 128);

  // ---- stage 2: sample_and_group(256, 32) + local_op(256) ----
  fps_kernel<<<Bn, 256, 0, stream>>>(nxyz1, S1, S2, fps2);
  gather_xyz<<<dim3((unsigned)((S2 * 3 + 255) / 256), Bn), 256, 0, stream>>>(nxyz1, fps2, nxyz2, S1, S2);
  gather_feat_bf<<<dim3((unsigned)((S2 * 128 + 255) / 256), Bn), 256, 0, stream>>>(f0bf, fps2, npts2, S1, S2, 128);
  knn_kernel<<<dim3(S2, Bn), 256, 0, stream>>>(nxyz1, nxyz2, S1, S2, 32, knn2);
  group_concat<<<g1d(COLS2 * 128), 256, 0, stream>>>(f0bf, npts2, knn2, bigA, S1, S2, 128, COLS2 * 128);
  gemm(w_lo1a, bigA, nullptr, bigB, ss_lo1a, 256, 256, COLS2, 1, 0, 0, 0, 0, 1);
  fill_zero_f32<<<g1d(PTS2 * 256), 256, 0, stream>>>(f1f, PTS2 * 256);
  gemm(w_lo1b, bigB, f1f, nullptr, ss_lo1b, 256, 256, COLS2, 1, /*maxpool*/1, 0, 0, 0, 1);
  cvt_f32_bf16<<<g1d(PTS2 * 256), 256, 0, stream>>>(f1f, f1bf, PTS2 * 256);

  // ---- transformer head ----
  conv3_kernel<<<g1d(PTS2 * 256), 256, 0, stream>>>(nxyz2, pos_W, nullptr, pos_b,
                                                    posb, nullptr, PTS2, 256, 0);
  gemm(w_pt, f1bf, cur0, nullptr, ss_pt, 256, 256, PTS2, 1, 0, 0, 0, 0, 1);

  const long long NE = PTS2 * 256;
  float* cur = cur0;
  for (int i = 0; i < 4; ++i) {
    ew_add2<<<g1d(NE), 256, 0, stream>>>(cur, posb, xin, xin_bf, NE);
    gemm(w_qk[i], xin_bf, nullptr, q_bf, ss_q[i], 64, 256, PTS2, 0, 0, 0, 0, 0, 1);
    gemm(w_v[i],  xin_bf, nullptr, v_bf, ss_v[i], 256, 256, PTS2, 0, 0, 0, 0, 0, 1);
    transpose_bf<<<dim3((unsigned)((256 * 256 + 255) / 256), Bn), 256, 0, stream>>>(v_bf, vT_bf, 256, 256);
    // energy[b][n][m] = sum_c q[n,c] q[m,c]   (batched, W == X == q)
    gemm(q_bf, q_bf, energy, nullptr, nullptr, 256, 64, 256, 0, 0,
         256 * 64, 256 * 64, 256 * 256, Bn);
    softmax_rows256<<<(unsigned)PTS2, 256, 0, stream>>>(energy);
    attn_finalize<<<dim3(256, Bn), 256, 0, stream>>>(energy, attnT);
    // xr[b][m][c] = sum_n vT[c,n] attnT[m,n]
    gemm(vT_bf, attnT, xr, nullptr, nullptr, 256, 256, 256, 0, 0,
         256 * 256, 256 * 256, 256 * 256, Bn);
    ew_sub_bf<<<g1d(NE), 256, 0, stream>>>(xin, xr, d_bf, NE);
    gemm(w_tr[i], d_bf, tbuf, nullptr, ss_tr[i], 256, 256, PTS2, /*leaky*/2, 0, 0, 0, 0, 1);
    ew_add2<<<g1d(NE), 256, 0, stream>>>(xin, tbuf, xbuf[i], nullptr, NE);
    cur = xbuf[i];
  }

  // ---- fuse + global max pool + final linear-bn ----
  concat_cat<<<g1d(PTS2 * 1280), 256, 0, stream>>>(xbuf[0], xbuf[1], xbuf[2], xbuf[3], f1f, cat_bf, PTS2);
  gemm(w_fuse, cat_bf, fused, nullptr, ss_fuse, 1024, 1280, PTS2, /*leaky*/2, 0, 0, 0, 0, 1);
  rowmax_pts<<<g1d((long long)Bn * 1024), 256, 0, stream>>>(fused, g_bf, S2, 1024, (long long)Bn * 1024);
  gemm(w_mlp2, g_bf, outp, nullptr, ss_fin, 512, 1024, Bn, 0, 0, 0, 0, 0, 1);
}
